// FusionModel_54958401519829
// MI455X (gfx1250) — matrix operand order repaired and checked by reading, not compiled, not run on hardware
//
#include <hip/hip_runtime.h>
#include <math.h>

// ---------------------------------------------------------------------------
// Multimodal fusion transformer for MI455X (gfx1250).
// bf16 WMMA GEMM core + double-buffered TDM weight-tile staging.
// ---------------------------------------------------------------------------

typedef __attribute__((ext_vector_type(16))) __bf16 v16bf;
typedef __attribute__((ext_vector_type(8)))  float  v8f;
typedef unsigned int u32x4 __attribute__((ext_vector_type(4)));
typedef int          i32x4 __attribute__((ext_vector_type(4)));
typedef int          i32x8 __attribute__((ext_vector_type(8)));

static constexpr int  Bsz = 8192;
static constexpr int  Dm  = 256;
static constexpr long BD  = (long)Bsz * Dm;

__device__ __forceinline__ unsigned int f2bf1(float f) {
  // round-to-nearest-even f32 -> bf16 (bits in low 16)
  unsigned int u = __float_as_uint(f);
  unsigned int r = u + 0x7FFFu + ((u >> 16) & 1u);
  return r >> 16;
}

union FragAB { v16bf v; uint4 q[2]; };

#if __has_builtin(__builtin_amdgcn_tensor_load_to_lds)
#define HAVE_TDM 1
// ---------------------------------------------------------------------------
// TDM: DMA a [rows=128 x cols=64] bf16 tile (row stride K elements) from
// global into LDS at byte offset ldsOff, inserting 16B of padding after each
// 128B row (pad_interval code 4 = 32 DWORDs, pad_amount code 3 = 4 DWORDs)
// so LDS rows land on a 144B (72-half) stride -> conflict-free b128 reads.
// Descriptor layout per CDNA5 ISA section 8.3-8.6.
// ---------------------------------------------------------------------------
__device__ __forceinline__ void tdm_load_b_tile(const unsigned short* gsrc,
                                                unsigned ldsOff, int K)
{
  unsigned long long ga = (unsigned long long)(size_t)gsrc;
  u32x4 g0;
  g0.x = 1u;                                              // count=1 (valid D#)
  g0.y = ldsOff;                                          // lds_addr (bytes)
  g0.z = (unsigned)ga;                                    // global_addr[31:0]
  g0.w = ((unsigned)(ga >> 32) & 0x1FFFFFFu) | (2u << 30);// addr[56:32]|type=2
  i32x8 g1;
  g1[0] = (1 << 16)      // data_size = 1 -> 2 bytes
        | (1 << 20)      // pad_enable
        | (4 << 22)      // pad_interval: 32 DWORDs (128B) before padding
        | (3 << 25);     // pad_amount: 4 DWORDs (16B)
  g1[1] = (K & 0xFFFF) << 16;                       // tensor_dim0[15:0]
  g1[2] = ((K >> 16) & 0xFFFF);                     // tensor_dim0[31:16], dim1 lo=0
  g1[3] = 16 | (64 << 16);                          // tensor_dim1 hi (=1M rows), tile_dim0=64
  g1[4] = 128;                                      // tile_dim1=128 rows, tile_dim2=0
  g1[5] = K;                                        // tensor_dim0_stride[31:0]
  g1[6] = 0;                                        // stride hi, dim1_stride lo
  g1[7] = 0;
  i32x4 g2 = (i32x4)0;                              // 2D: dims 2/3 unused
  i32x4 g3 = (i32x4)0;
#if __clang_major__ >= 23
  i32x8 gz = (i32x8)0;
  __builtin_amdgcn_tensor_load_to_lds(g0, g1, g2, g3, gz, 0);
#else
  __builtin_amdgcn_tensor_load_to_lds(g0, g1, g2, g3, 0);
#endif
}

__device__ __forceinline__ void tdm_wait0() {
#if __has_builtin(__builtin_amdgcn_s_wait_tensorcnt)
  __builtin_amdgcn_s_wait_tensorcnt(0);
#else
  asm volatile("s_wait_tensorcnt 0x0" ::: "memory");
#endif
}
#endif  // HAVE_TDM

// ---------------------------------------------------------------------------
// GEMM: C[M x N](f32) = A[M x K](f32, lda) * Bt^T  where Bt is bf16 [N x K]
// block tile 64(M) x 128(N), BK=64, 128 threads = 4 waves (wave grid 2x2),
// each wave computes 32x64 = 2x4 wmma tiles. grid = (N/128, M/64).
// Double-buffered: TDM DMA of B tile k+1 and global loads of A tile k+1
// overlap WMMA compute on tile k; TENSORcnt waited only before the barrier.
// gelu: 1 -> bias + exact GELU, 0 -> bias only.
// ---------------------------------------------------------------------------
__global__ void __launch_bounds__(128)
gemm_bf16_wmma(const float* __restrict__ A, int lda,
               const unsigned short* __restrict__ Bt,
               const float* __restrict__ bias,
               float* __restrict__ C, int ldc,
               int K, int gelu)
{
  __shared__ unsigned short As[2][64][72];    // 2 x 64 x (64+8 pad) bf16
  __shared__ unsigned short Bs[2][128][72];   // 2 x 128 x (64+8 pad) bf16

  const int t    = threadIdx.x;
  const int lane = t & 31;
  const int wave = t >> 5;
  const int wm   = (wave >> 1) << 5;          // 0 or 32
  const int wn   = (wave & 1)  << 6;          // 0 or 64
  const long blockM = (long)blockIdx.y << 6;
  const long blockN = (long)blockIdx.x << 7;

  v8f acc[2][4];
  const v8f vzero = {0.f,0.f,0.f,0.f,0.f,0.f,0.f,0.f};
#pragma unroll
  for (int i = 0; i < 2; i++)
#pragma unroll
    for (int j = 0; j < 4; j++) acc[i][j] = vzero;

  const float*          Ab = A  + blockM * (long)lda;
  const unsigned short* Bb = Bt + blockN * (long)K;
  const int nk = K >> 6;

  // stage A tile ki into buffer p: 64 rows x 64 f32 -> bf16
  auto stageA = [&](int ki, int p) {
    const int kt = ki << 6;
#pragma unroll
    for (int i = 0; i < 4; i++) {
      int u = t + (i << 7);
      int row = u >> 3, ch = u & 7;
      const float4* ap = (const float4*)(Ab + (long)row * lda + kt + (ch << 3));
      float4 f0 = ap[0], f1 = ap[1];
      uint4 pk;
      pk.x = f2bf1(f0.x) | (f2bf1(f0.y) << 16);
      pk.y = f2bf1(f0.z) | (f2bf1(f0.w) << 16);
      pk.z = f2bf1(f1.x) | (f2bf1(f1.y) << 16);
      pk.w = f2bf1(f1.z) | (f2bf1(f1.w) << 16);
      *(uint4*)&As[p][row][ch << 3] = pk;
    }
  };

#if !defined(HAVE_TDM)
  // fallback manual B staging: 128 rows x 64 bf16
  auto stageB = [&](int ki, int p) {
    const int kt = ki << 6;
#pragma unroll
    for (int i = 0; i < 8; i++) {
      int u = t + (i << 7);
      int row = u >> 3, ch = u & 7;
      *(uint4*)&Bs[p][row][ch << 3] =
          *(const uint4*)(Bb + (long)row * K + kt + (ch << 3));
    }
  };
#endif

  // compute on buffer p: 2 K-steps of 32 -> 16 WMMAs
  auto compute = [&](int p) {
#pragma unroll
    for (int kk = 0; kk < 64; kk += 32) {
      FragAB a[2], b[4];
      // A 16x32 frag: lanes<16 -> K {kk+0..7, kk+16..23}; lanes>=16 -> +8
      const int ak = kk + ((lane >> 4) << 3);
      // B 32x16 frag: lane n holds column n; lanes<16 K {kk+0..7, kk+16..23}; lanes>=16 -> +8
      const int bk = kk + ((lane >> 4) << 3);
#pragma unroll
      for (int tm = 0; tm < 2; tm++) {
        int r = wm + (tm << 4) + (lane & 15);
        a[tm].q[0] = *(const uint4*)&As[p][r][ak];
        a[tm].q[1] = *(const uint4*)&As[p][r][ak + 16];
      }
#pragma unroll
      for (int tn = 0; tn < 4; tn++) {
        int r = wn + (tn << 4) + (lane & 15);
        b[tn].q[0] = *(const uint4*)&Bs[p][r][bk];
        b[tn].q[1] = *(const uint4*)&Bs[p][r][bk + 16];
      }
#pragma unroll
      for (int tm = 0; tm < 2; tm++)
#pragma unroll
        for (int tn = 0; tn < 4; tn++)
          acc[tm][tn] = __builtin_amdgcn_wmma_f32_16x16x32_bf16(
              false, a[tm].v, false, b[tn].v, (short)0, acc[tm][tn],
              false, false);
    }
  };

  // ---- prologue: stage tile 0 into buffer 0 ----
#if defined(HAVE_TDM)
  if (t < 32) tdm_load_b_tile(Bb, (unsigned)(size_t)&Bs[0][0][0], K);
#else
  stageB(0, 0);
#endif
  stageA(0, 0);
#if defined(HAVE_TDM)
  if (t < 32) tdm_wait0();
#endif
  __syncthreads();

  // ---- main pipeline ----
  for (int i = 0; i < nk; i++) {
    const int p = i & 1;
    if (i + 1 < nk) {
#if defined(HAVE_TDM)
      // DMA next B tile while computing this one (EXEC ignored; wave 0 only)
      if (t < 32)
        tdm_load_b_tile(Bb + ((long)(i + 1) << 6),
                        (unsigned)(size_t)&Bs[p ^ 1][0][0], K);
#else
      stageB(i + 1, p ^ 1);
#endif
      stageA(i + 1, p ^ 1);
      if (i + 2 < nk)
        __builtin_prefetch(Ab + (long)(t >> 1) * lda + ((long)(i + 2) << 6) +
                               (t & 1) * 32, 0, 1);
    }
    compute(p);
#if defined(HAVE_TDM)
    if (t < 32) tdm_wait0();   // retire DMA of tile i+1 before barrier
#endif
    __syncthreads();
  }

  // epilogue: C/D layout -> lane n = col, vgpr i: row = i + (lane>=16)*8
#pragma unroll
  for (int tm = 0; tm < 2; tm++) {
    long grow0 = blockM + wm + (tm << 4) + ((lane >> 4) << 3);
#pragma unroll
    for (int tn = 0; tn < 4; tn++) {
      long gcol = blockN + wn + (tn << 4) + (lane & 15);
      float bv = bias ? bias[gcol] : 0.0f;
#pragma unroll
      for (int i = 0; i < 8; i++) {
        float v = acc[tm][tn][i] + bv;
        if (gelu) v = 0.5f * v * (1.0f + erff(v * 0.70710678118654752f));
        C[(grow0 + i) * (long)ldc + gcol] = v;
      }
    }
  }
}

// ---------------------------------------------------------------------------
// weight transpose f32 [K][N] -> bf16 [N][K]; blockIdx.z selects matrix
// ---------------------------------------------------------------------------
__global__ void __launch_bounds__(256)
transpose_w(const float* __restrict__ src, unsigned short* __restrict__ dst,
            int K, int N)
{
  long mat = blockIdx.z;
  src += mat * (long)K * N;
  dst += mat * (long)K * N;
  long idx = (long)blockIdx.x * 256 + threadIdx.x;   // < K*N
  int k = (int)(idx / N);
  int n = (int)(idx - (long)k * N);
  dst[(long)n * K + k] = (unsigned short)f2bf1(src[idx]);
}

// ---------------------------------------------------------------------------
// tokens[m][b][d] = emb_m[b][d] + mod_type[m][d]
// ---------------------------------------------------------------------------
__global__ void __launch_bounds__(256)
init_tokens(const float* __restrict__ e0, const float* __restrict__ e1,
            const float* __restrict__ e2, const float* __restrict__ e3,
            const float* __restrict__ e4, const float* __restrict__ mod,
            float* __restrict__ tok)
{
  long idx = (long)blockIdx.x * 256 + threadIdx.x;   // < 5*B*D
  long m = idx / BD;
  long r = idx - m * BD;
  int d  = (int)(r & 255);
  const float* e = (m == 0) ? e0 : (m == 1) ? e1 : (m == 2) ? e2
                 : (m == 3) ? e3 : e4;
  tok[idx] = e[r] + mod[m * 256 + d];
}

// ---------------------------------------------------------------------------
// attention: one thread per (b, h); 4 keys, DH=32, scale 1/sqrt(32)
// ---------------------------------------------------------------------------
__global__ void __launch_bounds__(256)
attn_kernel(const float* __restrict__ q, const float* __restrict__ k,
            const float* __restrict__ v, float* __restrict__ o)
{
  long idx = (long)blockIdx.x * 256 + threadIdx.x;   // < B*8
  long b = idx >> 3;
  int  h = (int)(idx & 7);
  const float4* qp = (const float4*)(q + b * 256 + h * 32);
  float4 qv[8];
#pragma unroll
  for (int i = 0; i < 8; i++) qv[i] = qp[i];
  float s[4];
#pragma unroll
  for (int j = 0; j < 4; j++) {
    const float4* kp = (const float4*)(k + ((long)j * Bsz + b) * 256 + h * 32);
    float acc = 0.0f;
#pragma unroll
    for (int i = 0; i < 8; i++) {
      float4 kv4 = kp[i];
      acc += qv[i].x * kv4.x + qv[i].y * kv4.y + qv[i].z * kv4.z + qv[i].w * kv4.w;
    }
    s[j] = acc * 0.17677669529663687f;   // 1/sqrt(32)
  }
  float mx = fmaxf(fmaxf(s[0], s[1]), fmaxf(s[2], s[3]));
  float e[4], se = 0.0f;
#pragma unroll
  for (int j = 0; j < 4; j++) { e[j] = expf(s[j] - mx); se += e[j]; }
  float inv = 1.0f / se;
  float4 ov[8];
#pragma unroll
  for (int i = 0; i < 8; i++) ov[i] = make_float4(0.f, 0.f, 0.f, 0.f);
#pragma unroll
  for (int j = 0; j < 4; j++) {
    float a = e[j] * inv;
    const float4* vp = (const float4*)(v + ((long)j * Bsz + b) * 256 + h * 32);
#pragma unroll
    for (int i = 0; i < 8; i++) {
      float4 vv = vp[i];
      ov[i].x += a * vv.x; ov[i].y += a * vv.y;
      ov[i].z += a * vv.z; ov[i].w += a * vv.w;
    }
  }
  float4* op = (float4*)(o + b * 256 + h * 32);
#pragma unroll
  for (int i = 0; i < 8; i++) op[i] = ov[i];
}

// ---------------------------------------------------------------------------
// wave-per-row layernorm: out = LN(res + g*y) * sc + bi
// y may be null (g unused); gatep null -> g = 1 else clamp(gatep[gidx],0,1)
// ---------------------------------------------------------------------------
__global__ void __launch_bounds__(256)
ln_rows(const float* __restrict__ res, const float* __restrict__ y,
        const float* __restrict__ gatep, int gidx,
        const float* __restrict__ sc, const float* __restrict__ bi,
        float* __restrict__ out, int Dlen, long nrows)
{
  long row = (long)blockIdx.x * 8 + (threadIdx.x >> 5);
  if (row >= nrows) return;
  int lane = threadIdx.x & 31;
  int nc = Dlen >> 5;
  float g = 1.0f;
  if (gatep) g = fminf(fmaxf(gatep[gidx], 0.0f), 1.0f);
  const float* rp = res + row * (long)Dlen;
  const float* yp = y ? (y + row * (long)Dlen) : nullptr;
  float sum = 0.0f, sq = 0.0f;
  for (int c = 0; c < nc; c++) {
    int d = lane + (c << 5);
    float vv = rp[d];
    if (yp) vv += g * yp[d];
    sum += vv; sq += vv * vv;
  }
  for (int o2 = 16; o2 > 0; o2 >>= 1) {
    sum += __shfl_xor(sum, o2, 32);
    sq  += __shfl_xor(sq,  o2, 32);
  }
  float mu  = sum / (float)Dlen;
  float var = sq / (float)Dlen - mu * mu;
  float rs  = rsqrtf(var + 1e-5f);
  float* op = out + row * (long)Dlen;
  for (int c = 0; c < nc; c++) {
    int d = lane + (c << 5);
    float vv = rp[d];
    if (yp) vv += g * yp[d];
    op[d] = (vv - mu) * rs * sc[d] + bi[d];
  }
}

// ---------------------------------------------------------------------------
// LN over gathered rows from normed[5][B][256]:
//   mode 1 (joint, D=1280): col d -> normed[d>>8][b][d&255]
//   mode 2 (affective, D=512): col d -> normed[d<256?2:4][b][d&255]
// ---------------------------------------------------------------------------
__device__ __forceinline__ float ln_gather_ld(const float* __restrict__ normed,
                                              long row, int d, int mode)
{
  int m = (mode == 1) ? (d >> 8) : ((d < 256) ? 2 : 4);
  int dd = d & 255;
  return normed[((long)m * Bsz + row) * 256 + dd];
}

__global__ void __launch_bounds__(256)
ln_gather(const float* __restrict__ normed,
          const float* __restrict__ sc, const float* __restrict__ bi,
          float* __restrict__ out, int Dlen, int mode)
{
  long row = (long)blockIdx.x * 8 + (threadIdx.x >> 5);
  if (row >= Bsz) return;
  int lane = threadIdx.x & 31;
  int nc = Dlen >> 5;
  float sum = 0.0f, sq = 0.0f;
  for (int c = 0; c < nc; c++) {
    float vv = ln_gather_ld(normed, row, lane + (c << 5), mode);
    sum += vv; sq += vv * vv;
  }
  for (int o2 = 16; o2 > 0; o2 >>= 1) {
    sum += __shfl_xor(sum, o2, 32);
    sq  += __shfl_xor(sq,  o2, 32);
  }
  float mu  = sum / (float)Dlen;
  float var = sq / (float)Dlen - mu * mu;
  float rs  = rsqrtf(var + 1e-5f);
  float* op = out + row * (long)Dlen;
  for (int c = 0; c < nc; c++) {
    int d = lane + (c << 5);
    float vv = ln_gather_ld(normed, row, d, mode);
    op[d] = (vv - mu) * rs * sc[d] + bi[d];
  }
}

// ---------------------------------------------------------------------------
// tiny head: out[b][0..3] = h[b][:] @ W[K][4] + bias
// ---------------------------------------------------------------------------
__global__ void __launch_bounds__(256)
head_out(const float* __restrict__ h, int K, const float* __restrict__ W,
         const float* __restrict__ bias, float* __restrict__ out)
{
  long b = (long)blockIdx.x * 256 + threadIdx.x;
  if (b >= Bsz) return;
  float a0 = bias[0], a1 = bias[1], a2 = bias[2], a3 = bias[3];
  const float* hp = h + b * (long)K;
  for (int k = 0; k < K; k++) {
    float x = hp[k];
    a0 += x * W[4 * k + 0];
    a1 += x * W[4 * k + 1];
    a2 += x * W[4 * k + 2];
    a3 += x * W[4 * k + 3];
  }
  out[b * 4 + 0] = a0; out[b * 4 + 1] = a1;
  out[b * 4 + 2] = a2; out[b * 4 + 3] = a3;
}

// ---------------------------------------------------------------------------
// host driver
// ---------------------------------------------------------------------------
extern "C" void kernel_launch(void* const* d_in, const int* in_sizes, int n_in,
                              void* d_out, int out_size, void* d_ws, size_t ws_size,
                              hipStream_t stream)
{
  (void)in_sizes; (void)n_in; (void)out_size; (void)ws_size;

  const float* emb0 = (const float*)d_in[0];
  const float* emb1 = (const float*)d_in[1];
  const float* emb2 = (const float*)d_in[2];
  const float* emb3 = (const float*)d_in[3];
  const float* emb4 = (const float*)d_in[4];
  const float* p_mod  = (const float*)d_in[5];
  const float* p_Wq   = (const float*)d_in[6];
  const float* p_bq   = (const float*)d_in[7];
  const float* p_Wk   = (const float*)d_in[8];
  const float* p_bk   = (const float*)d_in[9];
  const float* p_Wv   = (const float*)d_in[10];
  const float* p_bv   = (const float*)d_in[11];
  const float* p_Wo   = (const float*)d_in[12];
  const float* p_bo   = (const float*)d_in[13];
  const float* p_n1s  = (const float*)d_in[14];
  const float* p_n1b  = (const float*)d_in[15];
  const float* p_W1   = (const float*)d_in[16];
  const float* p_b1   = (const float*)d_in[17];
  const float* p_W2   = (const float*)d_in[18];
  const float* p_b2   = (const float*)d_in[19];
  const float* p_n2s  = (const float*)d_in[20];
  const float* p_n2b  = (const float*)d_in[21];
  const float* p_gate = (const float*)d_in[22];
  const float* p_ns   = (const float*)d_in[23];
  const float* p_nb   = (const float*)d_in[24];
  const float* p_tls  = (const float*)d_in[25];
  const float* p_tlb  = (const float*)d_in[26];
  const float* p_tW1  = (const float*)d_in[27];
  const float* p_tb1  = (const float*)d_in[28];
  const float* p_tW2  = (const float*)d_in[29];
  const float* p_tb2  = (const float*)d_in[30];
  const float* p_els  = (const float*)d_in[31];
  const float* p_elb  = (const float*)d_in[32];
  const float* p_eW1  = (const float*)d_in[33];
  const float* p_eb1  = (const float*)d_in[34];
  const float* p_eW2  = (const float*)d_in[35];
  const float* p_eb2  = (const float*)d_in[36];

  char* base = (char*)d_ws;
  size_t off = 0;
  auto take = [&](size_t bytes) -> void* {
    void* p = base + off;
    off += (bytes + 255) & ~(size_t)255;
    return p;
  };

  // bf16 transposed weights
  unsigned short* wqT  = (unsigned short*)take(10L * 65536 * 2);
  unsigned short* wkT  = (unsigned short*)take(10L * 65536 * 2);
  unsigned short* wvT  = (unsigned short*)take(10L * 65536 * 2);
  unsigned short* woT  = (unsigned short*)take(10L * 65536 * 2);
  unsigned short* w1T  = (unsigned short*)take(10L * 262144 * 2);
  unsigned short* w2T  = (unsigned short*)take(10L * 262144 * 2);
  unsigned short* tw1T = (unsigned short*)take(655360L * 2);
  unsigned short* ew1T = (unsigned short*)take(131072L * 2);

  // f32 activations
  float* tok0 = (float*)take(5 * BD * 4);
  float* tok1 = (float*)take(5 * BD * 4);
  float* qb   = (float*)take(BD * 4);
  float* kb   = (float*)take(4 * BD * 4);   // also reused as tj / ej
  float* vb   = (float*)take(4 * BD * 4);   // contiguous after kb
  float* ob   = (float*)take(BD * 4);
  float* ab   = (float*)take(BD * 4);
  float* xb   = (float*)take(BD * 4);
  float* h1b  = (float*)take((long)Bsz * 1024 * 4);
  float* fb   = (float*)take(BD * 4);
  (void)vb;

  // ---- weight prep (transpose + bf16) ----
  transpose_w<<<dim3(256, 1, 10), 256, 0, stream>>>(p_Wq, wqT, 256, 256);
  transpose_w<<<dim3(256, 1, 10), 256, 0, stream>>>(p_Wk, wkT, 256, 256);
  transpose_w<<<dim3(256, 1, 10), 256, 0, stream>>>(p_Wv, wvT, 256, 256);
  transpose_w<<<dim3(256, 1, 10), 256, 0, stream>>>(p_Wo, woT, 256, 256);
  transpose_w<<<dim3(1024, 1, 10), 256, 0, stream>>>(p_W1, w1T, 256, 1024);
  transpose_w<<<dim3(1024, 1, 10), 256, 0, stream>>>(p_W2, w2T, 1024, 256);
  transpose_w<<<dim3(2560, 1, 1), 256, 0, stream>>>(p_tW1, tw1T, 1280, 512);
  transpose_w<<<dim3(512, 1, 1), 256, 0, stream>>>(p_eW1, ew1T, 512, 256);

  // ---- tokens = emb + mod_type ----
  init_tokens<<<(5 * BD) / 256, 256, 0, stream>>>(emb0, emb1, emb2, emb3, emb4,
                                                  p_mod, tok0);

  static const int OTH[5][4] = {
      {1, 2, 3, 4}, {0, 2, 3, 4}, {0, 1, 3, 4}, {0, 1, 2, 4}, {0, 1, 2, 3}};

  const dim3 gN256(256 / 128, Bsz / 64);    // (2, 128)
  const dim3 gN1024(1024 / 128, Bsz / 64);  // (8, 128)
  const dim3 gN512(512 / 128, Bsz / 64);    // (4, 128)

  float* cur = tok0;
  float* nxt = tok1;
  for (int l = 0; l < 2; l++) {
    for (int m = 0; m < 5; m++) {
      const int lm = l * 5 + m;
      // Q projection
      gemm_bf16_wmma<<<gN256, 128, 0, stream>>>(
          cur + (long)m * BD, 256, wqT + (long)lm * 65536,
          p_bq + (long)lm * 256, qb, 256, 256, 0);
      // K/V projections of the 4 other modalities
      for (int j = 0; j < 4; j++) {
        const float* src = cur + (long)OTH[m][j] * BD;
        gemm_bf16_wmma<<<gN256, 128, 0, stream>>>(
            src, 256, wkT + (long)lm * 65536, p_bk + (long)lm * 256,
            kb + (long)j * BD, 256, 256, 0);
        gemm_bf16_wmma<<<gN256, 128, 0, stream>>>(
            src, 256, wvT + (long)lm * 65536, p_bv + (long)lm * 256,
            vb + (long)j * BD, 256, 256, 0);
      }
      // softmax attention over 4 keys
      attn_kernel<<<(Bsz * 8) / 256, 256, 0, stream>>>(qb, kb, vb, ob);
      // output projection
      gemm_bf16_wmma<<<gN256, 128, 0, stream>>>(
          ob, 256, woT + (long)lm * 65536, p_bo + (long)lm * 256, ab, 256,
          256, 0);
      // x = LN(tokens + clip(gate)*attended)
      ln_rows<<<Bsz / 8, 256, 0, stream>>>(
          cur + (long)m * BD, ab, p_gate, lm, p_n1s + (long)lm * 256,
          p_n1b + (long)lm * 256, xb, 256, Bsz);
      // FFN: h1 = gelu(x@W1 + b1)
      gemm_bf16_wmma<<<gN1024, 128, 0, stream>>>(
          xb, 256, w1T + (long)lm * 262144, p_b1 + (long)lm * 1024, h1b,
          1024, 256, 1);
      // ffn = h1@W2 + b2
      gemm_bf16_wmma<<<gN256, 128, 0, stream>>>(
          h1b, 1024, w2T + (long)lm * 262144, p_b2 + (long)lm * 256, fb, 256,
          1024, 0);
      // tokens_next = LN(x + ffn)
      ln_rows<<<Bsz / 8, 256, 0, stream>>>(
          xb, fb, nullptr, 0, p_n2s + (long)lm * 256, p_n2b + (long)lm * 256,
          nxt + (long)m * BD, 256, Bsz);
    }
    float* tmp = cur; cur = nxt; nxt = tmp;
  }
  // after 2 layers: cur == tok0 holds final tokens; normed -> tok1
  ln_rows<<<(5 * Bsz) / 8, 256, 0, stream>>>(cur, nullptr, nullptr, 0, p_ns,
                                             p_nb, tok1, 256, 5 * Bsz);
  float* normed = tok1;

  // tumor head: tj = LN(joint) -> gelu(tj@tW1+tb1) -> @tW2+tb2
  float* tj = kb;  // B x 1280 f32 (spans kb and start of vb; both done)
  ln_gather<<<Bsz / 8, 256, 0, stream>>>(normed, p_tls, p_tlb, tj, 1280, 1);
  gemm_bf16_wmma<<<gN512, 128, 0, stream>>>(tj, 1280, tw1T, p_tb1, h1b, 512,
                                            1280, 1);
  head_out<<<Bsz / 256, 256, 0, stream>>>(h1b, 512, p_tW2, p_tb2,
                                          (float*)d_out);

  // emotion head: ej = LN(concat(face, eeg_emotion)) -> gelu(@eW1) -> @eW2
  float* ejb = kb;  // B x 512 f32 (tj no longer needed)
  ln_gather<<<Bsz / 8, 256, 0, stream>>>(normed, p_els, p_elb, ejb, 512, 2);
  gemm_bf16_wmma<<<gN256, 128, 0, stream>>>(ejb, 512, ew1T, p_eb1, qb, 256,
                                            512, 1);
  head_out<<<Bsz / 256, 256, 0, stream>>>(qb, 256, p_eW2, p_eb2,
                                          (float*)d_out + (long)Bsz * 4);
}
